// ScaledDotProductRelativeAttention_34703335751906
// MI455X (gfx1250) — matrix operand order repaired and checked by reading, not compiled, not run on hardware
//
#include <hip/hip_runtime.h>
#include <hip/hip_bf16.h>

// ---------------------------------------------------------------------------
// Relative-position attention, fused, for MI455X (gfx1250, wave32, WMMA).
// B=4 H=16 T=1024 D=64.  Pre-pass converts K -> bf16 [s][d] and V -> bf16
// transposed [d][s] into d_ws; the main kernel then stages chunks into LDS
// with CDNA5 async global->LDS copies (no conversion in the hot loop) and
// runs both GEMMs with v_wmma_f32_16x16x32_bf16 (fp32 accumulation).
// ---------------------------------------------------------------------------

#define B_ 4
#define H_ 16
#define T_ 1024
#define D_ 64
#define RELW 33          // 2*16+1
#define ROWS 32          // query rows per workgroup
#define NTHREADS 256     // 8 waves
#define SSTR 1028        // Ss row stride (floats): 16B-aligned, bank stride 4

typedef __attribute__((ext_vector_type(16))) __bf16 v16bf;
typedef __attribute__((ext_vector_type(8)))  __bf16 v8bf;
typedef __attribute__((ext_vector_type(8)))  float  v8f;
typedef int v4i __attribute__((vector_size(16)));   // matches builtin param type

union V16U { v16bf v; v8bf h[2]; };

#if defined(__HIP_DEVICE_COMPILE__) && defined(__has_builtin)
#if __has_builtin(__builtin_amdgcn_global_load_async_to_lds_b128) && \
    __has_builtin(__builtin_amdgcn_s_wait_asynccnt)
#define HAVE_ASYNC 1
#endif
#endif
#ifndef HAVE_ASYNC
#define HAVE_ASYNC 0
#endif

// float -> bf16 bits (round to nearest even)
static __device__ inline unsigned short f2bfbits(float f) {
    union { float f; unsigned u; } x; x.f = f;
    return (unsigned short)((x.u + 0x7FFFu + ((x.u >> 16) & 1u)) >> 16);
}
static __device__ inline __bf16 f2bf(float f) {
    union { unsigned short s; __bf16 b; } y; y.s = f2bfbits(f);
    return y.b;
}
static __device__ inline float bf2f(__bf16 b) {
    union { unsigned short s; __bf16 b; } y; y.b = b;
    union { float f; unsigned u; } x; x.u = ((unsigned)y.s) << 16;
    return x.f;
}

#if HAVE_ASYNC
// async 16-byte global->LDS copy (tracked by ASYNCcnt)
static __device__ inline void async_cp16(const __bf16* g, __bf16* l) {
    __builtin_amdgcn_global_load_async_to_lds_b128(
        (__attribute__((address_space(1))) v4i*)g,
        (__attribute__((address_space(3))) v4i*)l, 0, 0);
}
#endif

// ------------------- pre-pass: K -> bf16 [s][d]; V -> bf16 [d][s] -------------------
__global__ __launch_bounds__(NTHREADS) void
convert_kv(const float* __restrict__ k, const float* __restrict__ v,
           __bf16* __restrict__ kb, __bf16* __restrict__ vt)
{
    __shared__ float tile[64][65];
    const int tid   = threadIdx.x;
    const int bh    = blockIdx.y;
    const int chunk = blockIdx.x;          // 64-row slab of s
    const size_t base = (size_t)bh * T_ * D_;

    const int sl = tid >> 2;               // 0..63
    const int db = (tid & 3) * 16;         // 0..48

    // K: straight convert (coalesced both sides)
    {
        const float* kp = k + base + (size_t)(chunk * 64 + sl) * D_ + db;
        __bf16* op = kb + base + (size_t)(chunk * 64 + sl) * D_ + db;
        #pragma unroll
        for (int half = 0; half < 2; ++half) {
            float4 f0 = *(const float4*)(kp + half * 8);
            float4 f1 = *(const float4*)(kp + half * 8 + 4);
            v8bf o; o[0]=f2bf(f0.x); o[1]=f2bf(f0.y); o[2]=f2bf(f0.z); o[3]=f2bf(f0.w);
                    o[4]=f2bf(f1.x); o[5]=f2bf(f1.y); o[6]=f2bf(f1.z); o[7]=f2bf(f1.w);
            *(v8bf*)(op + half * 8) = o;
        }
    }
    // V: transpose via LDS, then convert (coalesced writes)
    {
        const float* vp = v + base + (size_t)(chunk * 64 + sl) * D_ + db;
        #pragma unroll
        for (int j = 0; j < 16; j += 4) {
            float4 f = *(const float4*)(vp + j);
            tile[sl][db + j + 0] = f.x; tile[sl][db + j + 1] = f.y;
            tile[sl][db + j + 2] = f.z; tile[sl][db + j + 3] = f.w;
        }
    }
    __syncthreads();
    {
        const int dl = tid >> 2;           // output row (d)
        const int sb = (tid & 3) * 16;     // s offset within chunk
        __bf16* op = vt + (size_t)bh * D_ * T_ + (size_t)dl * T_ + chunk * 64 + sb;
        #pragma unroll
        for (int half = 0; half < 2; ++half) {
            v8bf o;
            #pragma unroll
            for (int j = 0; j < 8; ++j)
                o[j] = f2bf(tile[sb + half * 8 + j][dl]);
            *(v8bf*)(op + half * 8) = o;
        }
    }
}

// ------------------- fused attention kernel -------------------
template <bool USE_WS>
__global__ __launch_bounds__(NTHREADS) void
rel_attn_fused(const float* __restrict__ q,
               const float* __restrict__ k,
               const float* __restrict__ v,
               const float* __restrict__ emb_k,
               const float* __restrict__ emb_v,
               float* __restrict__ out,
               float* __restrict__ attn,
               const __bf16* __restrict__ kb,
               const __bf16* __restrict__ vt)
{
    // ---- LDS (162,304 B total) ----
    __shared__ __align__(16) __bf16 Qs[ROWS][72];
    __shared__ float  Ps[ROWS][RELW];
    __shared__ __align__(16) float  Ss[ROWS][SSTR];
    __shared__ float  Ws[ROWS][RELW];
    __shared__ float  Ev[RELW][D_];
    __shared__ __align__(16) __bf16 stage[64 * 72];

    const int tid  = threadIdx.x;
    const int wave = tid >> 5;
    const int lane = tid & 31;
    const int hi   = lane >> 4;
    const int ln   = lane & 15;

    const int bh      = blockIdx.y;
    const int rowbase = blockIdx.x * ROWS;

    const size_t qk_base = (size_t)bh * T_ * D_;
    const float scale = 0.125f;            // 1/sqrt(64)

    // ---------------- Phase 0: stage Q (bf16) and emb_v ----------------
    {
        int row = tid >> 3;
        int db  = (tid & 7) * 8;
        const float* qp = q + qk_base + (size_t)(rowbase + row) * D_ + db;
        #pragma unroll
        for (int j = 0; j < 8; j += 4) {
            float4 f = *(const float4*)(qp + j);
            Qs[row][db + j + 0] = f2bf(f.x);
            Qs[row][db + j + 1] = f2bf(f.y);
            Qs[row][db + j + 2] = f2bf(f.z);
            Qs[row][db + j + 3] = f2bf(f.w);
        }
        for (int i = tid; i < RELW * D_; i += NTHREADS)
            Ev[i / D_][i % D_] = emb_v[i];
    }
    __syncthreads();

    // ---------------- Phase 1: P[t][r] = q[t] . emb_k[r] ----------------
    for (int i = tid; i < ROWS * RELW; i += NTHREADS) {
        int tl = i / RELW, r = i % RELW;
        float s = 0.f;
        #pragma unroll 8
        for (int d = 0; d < D_; ++d)
            s += bf2f(Qs[tl][d]) * emb_k[r * D_ + d];
        Ps[tl][r] = s;
    }

    // Per-wave A operand (Q rows): runs [8hi..+7],[16+8hi..+7] per ISA striping
    const int rb = wave >> 2;
    const int wc = wave & 3;
    V16U aq0, aq1;
    {
        const __bf16* qrow = &Qs[rb * 16 + ln][0];
        aq0.h[0] = *(const v8bf*)(qrow + 8 * hi);
        aq0.h[1] = *(const v8bf*)(qrow + 16 + 8 * hi);
        aq1.h[0] = *(const v8bf*)(qrow + 32 + 8 * hi);
        aq1.h[1] = *(const v8bf*)(qrow + 48 + 8 * hi);
    }

    // ---------------- Phase 2: logits = (QK^T + bias) * scale ----------------
    for (int ch = 0; ch < T_ / 64; ++ch) {
        __syncthreads();
        {   // stage K chunk row-major [s_local][d], bf16
            int sl = tid >> 2;
            int db = (tid & 3) * 16;
            if constexpr (USE_WS) {
                const __bf16* src = kb + qk_base + (size_t)(ch * 64 + sl) * D_ + db;
                if (ch < T_ / 64 - 1) __builtin_prefetch(src + 64 * D_, 0, 0);
#if HAVE_ASYNC
                async_cp16(src,     &stage[sl * 72 + db]);
                async_cp16(src + 8, &stage[sl * 72 + db + 8]);
                __builtin_amdgcn_s_wait_asynccnt(0);
#else
                *(v8bf*)&stage[sl * 72 + db]     = *(const v8bf*)src;
                *(v8bf*)&stage[sl * 72 + db + 8] = *(const v8bf*)(src + 8);
#endif
            } else {
                const float* kp = k + qk_base + (size_t)(ch * 64 + sl) * D_ + db;
                if (ch < T_ / 64 - 1) __builtin_prefetch(kp + 64 * D_, 0, 0);
                #pragma unroll
                for (int j = 0; j < 16; j += 4) {
                    float4 f = *(const float4*)(kp + j);
                    stage[sl * 72 + db + j + 0] = f2bf(f.x);
                    stage[sl * 72 + db + j + 1] = f2bf(f.y);
                    stage[sl * 72 + db + j + 2] = f2bf(f.z);
                    stage[sl * 72 + db + j + 3] = f2bf(f.w);
                }
            }
        }
        __syncthreads();

        V16U b0, b1;   // B striping: runs [8hi..+7],[16+8hi..+7] (matches A)
        {
            const __bf16* srow = &stage[(wc * 16 + ln) * 72];
            b0.h[0] = *(const v8bf*)(srow + 8 * hi);
            b0.h[1] = *(const v8bf*)(srow + 16 + 8 * hi);
            b1.h[0] = *(const v8bf*)(srow + 32 + 8 * hi);
            b1.h[1] = *(const v8bf*)(srow + 48 + 8 * hi);
        }
        v8f acc = {};
        acc = __builtin_amdgcn_wmma_f32_16x16x32_bf16(false, aq0.v, false, b0.v,
                                                      (short)0, acc, false, false);
        acc = __builtin_amdgcn_wmma_f32_16x16x32_bf16(false, aq1.v, false, b1.v,
                                                      (short)0, acc, false, false);

        const int scol = ch * 64 + wc * 16 + ln;
        #pragma unroll
        for (int e = 0; e < 8; ++e) {
            int m  = e + hi * 8;
            int tl = rb * 16 + m;
            int tg = rowbase + tl;
            int dd = scol - tg;
            dd = dd < -16 ? -16 : (dd > 16 ? 16 : dd);
            Ss[tl][scol] = (acc[e] + Ps[tl][dd + 16]) * scale;
        }
    }
    __syncthreads();

    // ---------------- Phase 3: softmax; attn -> global (f32) + LDS (packed bf16) ----------------
    for (int row = wave; row < ROWS; row += 8) {
        const int tg = rowbase + row;
        float m = -3.402823466e38f;
        for (int c = lane; c < T_; c += 32) m = fmaxf(m, Ss[row][c]);
        #pragma unroll
        for (int o = 16; o > 0; o >>= 1) m = fmaxf(m, __shfl_xor(m, o, 32));

        float sum = 0.f;
        for (int c = lane; c < T_; c += 32) {
            float e = __expf(Ss[row][c] - m);
            Ss[row][c] = e;
            sum += e;
        }
        #pragma unroll
        for (int o = 16; o > 0; o >>= 1) sum += __shfl_xor(sum, o, 32);
        const float inv = 1.f / sum;

        float w0 = 0.f, w32 = 0.f;
        float* arow = attn + ((size_t)bh * T_ + tg) * T_;
        unsigned* prow = (unsigned*)&Ss[row][0];   // packed bf16 overlay
        for (int c = lane; c < T_; c += 32) {
            float a = Ss[row][c] * inv;            // reads dwords [32i..32i+31]
            __builtin_nontemporal_store(a, arow + c);
            if (c <= tg - 16) w0  += a;
            if (c >= tg + 16) w32 += a;
            float an = __shfl_xor(a, 1, 32);
            if (!(lane & 1))                       // writes dwords [16i..16i+15]
                prow[c >> 1] = (unsigned)f2bfbits(a) | ((unsigned)f2bfbits(an) << 16);
        }
        #pragma unroll
        for (int o = 16; o > 0; o >>= 1) {
            w0  += __shfl_xor(w0,  o, 32);
            w32 += __shfl_xor(w32, o, 32);
        }
        if (lane == 0) { Ws[row][0] = w0; Ws[row][32] = w32; }
        if (lane >= 1) {
            int s = tg + lane - 16;
            const __bf16* pb = (const __bf16*)&Ss[row][0];
            Ws[row][lane] = (s >= 0 && s < T_) ? bf2f(pb[s]) : 0.f;
        }
    }

    // ---------------- Phase 4: out = attn*V + W*emb_v ----------------
    const int dt  = wave & 3;
    const int rb4 = wave >> 2;
    const __bf16* arow = (const __bf16*)&Ss[rb4 * 16 + ln][0];
    v8f oacc = {};
    for (int ch = 0; ch < T_ / 64; ++ch) {
        __syncthreads();
        {   // stage V chunk [d][s_local], bf16
            if constexpr (USE_WS) {
                int dl = tid >> 2;
                int sb = (tid & 3) * 16;
                const __bf16* src = vt + (size_t)bh * D_ * T_ + (size_t)dl * T_
                                       + ch * 64 + sb;
                if (ch < T_ / 64 - 1) __builtin_prefetch(src + 64, 0, 0);
#if HAVE_ASYNC
                async_cp16(src,     &stage[dl * 72 + sb]);
                async_cp16(src + 8, &stage[dl * 72 + sb + 8]);
                __builtin_amdgcn_s_wait_asynccnt(0);
#else
                *(v8bf*)&stage[dl * 72 + sb]     = *(const v8bf*)src;
                *(v8bf*)&stage[dl * 72 + sb + 8] = *(const v8bf*)(src + 8);
#endif
            } else {
                int sl = tid >> 2;
                int db = (tid & 3) * 16;
                const float* vp = v + qk_base + (size_t)(ch * 64 + sl) * D_ + db;
                if (ch < T_ / 64 - 1) __builtin_prefetch(vp + 64 * D_, 0, 0);
                #pragma unroll
                for (int j = 0; j < 16; j += 4) {
                    float4 f = *(const float4*)(vp + j);
                    stage[(db + j + 0) * 72 + sl] = f2bf(f.x);
                    stage[(db + j + 1) * 72 + sl] = f2bf(f.y);
                    stage[(db + j + 2) * 72 + sl] = f2bf(f.z);
                    stage[(db + j + 3) * 72 + sl] = f2bf(f.w);
                }
            }
        }
        __syncthreads();

        #pragma unroll
        for (int sub = 0; sub < 2; ++sub) {
            const int kb2 = ch * 64 + sub * 32;
            V16U a, b;
            a.h[0] = *(const v8bf*)(arow + kb2 + 8 * hi);
            a.h[1] = *(const v8bf*)(arow + kb2 + 16 + 8 * hi);
            const __bf16* vrow = &stage[(dt * 16 + ln) * 72];
            b.h[0] = *(const v8bf*)(vrow + sub * 32 + 8 * hi);
            b.h[1] = *(const v8bf*)(vrow + sub * 32 + 16 + 8 * hi);
            oacc = __builtin_amdgcn_wmma_f32_16x16x32_bf16(false, a.v, false, b.v,
                                                           (short)0, oacc, false, false);
        }
    }

    // epilogue: add W * emb_v and store
    #pragma unroll
    for (int e = 0; e < 8; ++e) {
        int m   = e + hi * 8;
        int tl  = rb4 * 16 + m;
        int dcc = dt * 16 + ln;
        float rel = 0.f;
        #pragma unroll
        for (int r = 0; r < RELW; ++r) rel += Ws[tl][r] * Ev[r][dcc];
        out[((size_t)bh * T_ + rowbase + tl) * D_ + dcc] = oacc[e] + rel;
    }
}

extern "C" void kernel_launch(void* const* d_in, const int* in_sizes, int n_in,
                              void* d_out, int out_size, void* d_ws, size_t ws_size,
                              hipStream_t stream) {
    const float* q     = (const float*)d_in[0];
    const float* k     = (const float*)d_in[1];
    const float* v     = (const float*)d_in[2];
    const float* emb_k = (const float*)d_in[3];
    const float* emb_v = (const float*)d_in[4];

    float* out  = (float*)d_out;                                // [B,H,T,D]
    float* attn = (float*)d_out + (size_t)B_ * H_ * T_ * D_;    // [B,H,T,T]

    dim3 grid(T_ / ROWS, B_ * H_, 1);
    dim3 block(NTHREADS, 1, 1);

    const size_t kv_elems = (size_t)B_ * H_ * T_ * D_;
    const size_t ws_need  = 2 * kv_elems * sizeof(__bf16);      // 16 MiB

    if (ws_size >= ws_need) {
        __bf16* kb = (__bf16*)d_ws;
        __bf16* vt = kb + kv_elems;
        convert_kv<<<dim3(T_ / 64, B_ * H_, 1), block, 0, stream>>>(k, v, kb, vt);
        rel_attn_fused<true><<<grid, block, 0, stream>>>(q, k, v, emb_k, emb_v,
                                                         out, attn, kb, vt);
    } else {
        rel_attn_fused<false><<<grid, block, 0, stream>>>(q, k, v, emb_k, emb_v,
                                                          out, attn, nullptr, nullptr);
    }
}
